// Shakespeare_33148557591139
// MI455X (gfx1250) — compile-verified, matrix operand order checked by reading, not run on hardware
//
#include <hip/hip_runtime.h>

// ---------------------------------------------------------------------------
// Types for CDNA5 WMMA (wave32, v_wmma_f32_16x16x32_bf16)
// ---------------------------------------------------------------------------
typedef __attribute__((ext_vector_type(16))) __bf16 v16bf;
typedef __attribute__((ext_vector_type(8)))  float  v8f;
typedef __attribute__((ext_vector_type(4)))  float  v4f;
typedef __attribute__((ext_vector_type(4)))  unsigned int v4u;
typedef __attribute__((ext_vector_type(4)))  int    v4i;

typedef __attribute__((address_space(1))) v4i gv4i;   // global int4
typedef __attribute__((address_space(3))) v4i lv4i;   // LDS int4

union F16frag {
    v4u q[2];
    v16bf v;
    unsigned short u[16];
};

__device__ __forceinline__ unsigned short f2bf(float f) {
    unsigned int u = __builtin_bit_cast(unsigned int, f);
    u += 0x7FFFu + ((u >> 16) & 1u);          // round-to-nearest-even
    return (unsigned short)(u >> 16);
}

// ---------------------------------------------------------------------------
// gfx1250 async global->LDS copy (ASYNCcnt path), with safe fallback.
// ---------------------------------------------------------------------------
#if __has_builtin(__builtin_amdgcn_global_load_async_to_lds_b128)
#define HAVE_ASYNC_LDS 1
#else
#define HAVE_ASYNC_LDS 0
#endif

__device__ __forceinline__ gv4i* to_as1(const void* p) {
    return (gv4i*)(unsigned long long)(uintptr_t)p;
}
__device__ __forceinline__ lv4i* to_as3(void* p) {
    // generic LDS-aperture address: low 32 bits are the LDS byte offset
    return (lv4i*)(unsigned)(uintptr_t)p;
}

// copy 32 bf16 (64B) global -> LDS
__device__ __forceinline__ void stage_a32(unsigned short* dst,
                                          const unsigned short* src) {
#if HAVE_ASYNC_LDS
#pragma unroll
    for (int i = 0; i < 4; ++i)
        __builtin_amdgcn_global_load_async_to_lds_b128(
            to_as1(src + i * 8), to_as3(dst + i * 8), 0, 0);
#else
#pragma unroll
    for (int i = 0; i < 4; ++i)
        ((v4u*)dst)[i] = ((const v4u*)src)[i];
#endif
}

__device__ __forceinline__ void wait_async0() {
#if HAVE_ASYNC_LDS
#if __has_builtin(__builtin_amdgcn_s_wait_asynccnt)
    __builtin_amdgcn_s_wait_asynccnt(0);
#else
    asm volatile("s_wait_asynccnt 0x0" ::: "memory");
#endif
#endif
}

// Load a 16x32 bf16 A/B fragment from an LDS tile stored [row][k] with
// row-stride `ld` (ushorts, multiple of 8). CDNA5 16-bit layout:
// lane r (0-15): K = kbase+[0..7] and kbase+[16..23]
// lane r+16   : K = kbase+[8..15] and kbase+[24..31]
__device__ __forceinline__ v16bf ld_frag(const unsigned short* tile, int ld,
                                         int rowbase, int kbase, int lane) {
    int r  = rowbase + (lane & 15);
    int kb = kbase + ((lane >> 4) << 3);
    const unsigned short* p = tile + r * ld + kb;
    F16frag f;
    f.q[0] = *(const v4u*)(p);
    f.q[1] = *(const v4u*)(p + 16);
    return f.v;
}

// pack 8 f32 -> 8 bf16 and store 16B to LDS
__device__ __forceinline__ void store_w8(unsigned short* wdst, v4f w0, v4f w1) {
    v4u o;
    o[0] = (unsigned)f2bf(w0[0]) | ((unsigned)f2bf(w0[1]) << 16);
    o[1] = (unsigned)f2bf(w0[2]) | ((unsigned)f2bf(w0[3]) << 16);
    o[2] = (unsigned)f2bf(w1[0]) | ((unsigned)f2bf(w1[1]) << 16);
    o[3] = (unsigned)f2bf(w1[2]) | ((unsigned)f2bf(w1[3]) << 16);
    *(v4u*)wdst = o;
}

// ---------------------------------------------------------------------------
// GEMM: out[M,N] = act(A[M,K] @ W[N,K]^T + bias) (+ residual)
// A: bf16 row-major. W: f32 row-major [N,K], converted to bf16 into LDS.
// OUT_MODE: 0 = f32 store, 1 = f32 store + residual add, 2 = bf16 store
// Block: 256 thr (8 waves). Tile: BM=256 x BN=64, K-step 32, double-buffered
// via async global->LDS for A and register-prefetch + cvt for W.
// Each wave owns two 16-row strips: 6 frag loads -> 8 WMMAs per K-step.
// ---------------------------------------------------------------------------
constexpr int BM = 256, BN = 64, BK = 32, LDT = 40;

template <int OUT_MODE, bool BIAS, bool RELU>
__global__ __launch_bounds__(256) void gemm_bf16w(
    const unsigned short* __restrict__ A, const float* __restrict__ W,
    const float* __restrict__ bias, float* __restrict__ outf,
    unsigned short* __restrict__ outb, const float* __restrict__ resid,
    int N, int K) {
    __shared__ unsigned short As[2][BM * LDT];
    __shared__ unsigned short Ws[2][BN * LDT];

    int tid = threadIdx.x;
    int lane = tid & 31, wave = tid >> 5;
    int n0 = blockIdx.x * BN;
    int m0 = blockIdx.y * BM;

    v8f acc[2][4];
#pragma unroll
    for (int s = 0; s < 2; ++s)
#pragma unroll
        for (int j = 0; j < 4; ++j)
#pragma unroll
            for (int v = 0; v < 8; ++v) acc[s][j][v] = 0.0f;

    int wrow = tid >> 2, wseg = (tid & 3) << 3;          // 64 rows x 32
    const unsigned short* abase = A + (size_t)(m0 + tid) * K;   // 1 row/thread
    const float* wbase = W + (size_t)(n0 + wrow) * K + wseg;

    // ---- prologue: stage k-step 0 into buffer 0 ----
    stage_a32(As[0] + tid * LDT, abase);
    {
        v4f w0 = *(const v4f*)(wbase);
        v4f w1 = *(const v4f*)(wbase + 4);
        store_w8(Ws[0] + wrow * LDT + wseg, w0, w1);
    }
    wait_async0();
    __syncthreads();

    int nsteps = K / BK;
    for (int step = 0; step < nsteps; ++step) {
        int cur = step & 1;
        bool more = (step + 1) < nsteps;
        int kn = (step + 1) * BK;

        v4f w0 = {0.f, 0.f, 0.f, 0.f}, w1 = {0.f, 0.f, 0.f, 0.f};
        if (more) {
            stage_a32(As[cur ^ 1] + tid * LDT, abase + kn);  // async copy
            w0 = *(const v4f*)(wbase + kn);                  // reg prefetch
            w1 = *(const v4f*)(wbase + kn + 4);
        }

        // compute on current buffers
        v16bf a0 = ld_frag(As[cur], LDT, wave * 32, 0, lane);
        v16bf a1 = ld_frag(As[cur], LDT, wave * 32 + 16, 0, lane);
#pragma unroll
        for (int j = 0; j < 4; ++j) {
            v16bf bf = ld_frag(Ws[cur], LDT, j * 16, 0, lane);
            acc[0][j] = __builtin_amdgcn_wmma_f32_16x16x32_bf16(
                false, a0, false, bf, (short)0, acc[0][j], false, false);
            acc[1][j] = __builtin_amdgcn_wmma_f32_16x16x32_bf16(
                false, a1, false, bf, (short)0, acc[1][j], false, false);
        }

        if (more) {
            store_w8(Ws[cur ^ 1] + wrow * LDT + wseg, w0, w1);
            wait_async0();
        }
        __syncthreads();
    }

    // ---- epilogue ----
    int nn = lane & 15;
    int mb = (lane >> 4) << 3;
#pragma unroll
    for (int s = 0; s < 2; ++s) {
        int mrow0 = m0 + wave * 32 + s * 16 + mb;
#pragma unroll
        for (int j = 0; j < 4; ++j) {
            int n = n0 + j * 16 + nn;
            float bv = 0.0f;
            if constexpr (BIAS) bv = bias[n];
#pragma unroll
            for (int v = 0; v < 8; ++v) {
                size_t idx = (size_t)(mrow0 + v) * (size_t)N + n;
                float val = acc[s][j][v] + bv;
                if constexpr (RELU) val = fmaxf(val, 0.0f);
                if constexpr (OUT_MODE == 1) val += resid[idx];
                if constexpr (OUT_MODE == 2) outb[idx] = f2bf(val);
                else                         outf[idx] = val;
            }
        }
    }
}

// ---------------------------------------------------------------------------
// Embedding: h[bt, d] = tok_emb[x[bt], d] + pos_emb[t, d]
// ---------------------------------------------------------------------------
__global__ __launch_bounds__(256) void embed_k(const int* __restrict__ x,
                                               const float* __restrict__ tok,
                                               const float* __restrict__ pos,
                                               float* __restrict__ h) {
    int row = blockIdx.x, tid = threadIdx.x;
    int t = row & 1023;                 // T = 1024
    int token = x[row];
#pragma unroll
    for (int i = 0; i < 3; ++i) {
        int d = tid + (i << 8);
        h[(size_t)row * 768 + d] =
            tok[(size_t)token * 768 + d] + pos[(size_t)t * 768 + d];
    }
}

// ---------------------------------------------------------------------------
// LayerNorm over D=768, one block per row, bf16 output.
// ---------------------------------------------------------------------------
__global__ __launch_bounds__(256) void ln_k(const float* __restrict__ x,
                                            const float* __restrict__ g,
                                            const float* __restrict__ b,
                                            unsigned short* __restrict__ y) {
    __shared__ float red[256];
    int row = blockIdx.x, tid = threadIdx.x;
    const float* xr = x + (size_t)row * 768;
    float v0 = xr[tid], v1 = xr[tid + 256], v2 = xr[tid + 512];
    red[tid] = v0 + v1 + v2;
    __syncthreads();
    for (int off = 128; off > 0; off >>= 1) {
        if (tid < off) red[tid] += red[tid + off];
        __syncthreads();
    }
    float mean = red[0] * (1.0f / 768.0f);
    __syncthreads();
    float d0 = v0 - mean, d1 = v1 - mean, d2 = v2 - mean;
    red[tid] = d0 * d0 + d1 * d1 + d2 * d2;
    __syncthreads();
    for (int off = 128; off > 0; off >>= 1) {
        if (tid < off) red[tid] += red[tid + off];
        __syncthreads();
    }
    float rstd = rsqrtf(red[0] * (1.0f / 768.0f) + 1e-5f);
    unsigned short* yr = y + (size_t)row * 768;
    yr[tid]       = f2bf(d0 * rstd * g[tid]       + b[tid]);
    yr[tid + 256] = f2bf(d1 * rstd * g[tid + 256] + b[tid + 256]);
    yr[tid + 512] = f2bf(d2 * rstd * g[tid + 512] + b[tid + 512]);
}

// ---------------------------------------------------------------------------
// Causal flash attention. One wave per (b, h, 16-query tile).
// qkv: bf16 [B*T, 2304] (q|k|v per row). xatt: bf16 [B*T, 768], head-merged.
// Keys processed 32 at a time: two 16x16 score subtiles per chunk, dense
// K=32 PV WMMAs, row max shuffles amortized over 32 keys, and row sums from
// a 5th WMMA accumulator against an all-ones B fragment.
// ---------------------------------------------------------------------------
__global__ __launch_bounds__(32) void attn_k(
    const unsigned short* __restrict__ qkv, unsigned short* __restrict__ xatt) {
    constexpr int T = 1024, H = 12, HD = 64, C3 = 2304, D = 768;
    __shared__ unsigned short qt_[16 * 72];
    __shared__ unsigned short kt_[32 * 72];
    __shared__ unsigned short vt_[32 * 72];
    __shared__ unsigned short pb_[16 * 40];

    int lane = threadIdx.x;
    int qt = blockIdx.x & 63;               // T/16 = 64
    int hh = (blockIdx.x >> 6) % H;
    int bb = blockIdx.x / (64 * H);

    {   // load Q tile 16x64
        int r = lane & 15, half = lane >> 4;
        const v4u* src = (const v4u*)(qkv +
            (size_t)(bb * T + qt * 16 + r) * C3 + hh * HD + half * 32);
        v4u* dst = (v4u*)(qt_ + r * 72 + half * 32);
#pragma unroll
        for (int i = 0; i < 4; ++i) dst[i] = src[i];
    }
    __syncthreads();
    v16bf qf0 = ld_frag(qt_, 72, 0, 0, lane);
    v16bf qf1 = ld_frag(qt_, 72, 0, 32, lane);

    // constant all-ones B fragment (all 32 K rows; masked keys carry P=0)
    F16frag ones;
#pragma unroll
    for (int i = 0; i < 16; ++i) ones.u[i] = 0x3F80u;

    v8f o[5];                                // o[0..3]=PV, o[4]=row sums
#pragma unroll
    for (int j = 0; j < 5; ++j)
#pragma unroll
        for (int v = 0; v < 8; ++v) o[j][v] = 0.0f;
    float mst[8];
#pragma unroll
    for (int v = 0; v < 8; ++v) mst[v] = -3.0e38f;

    int nn = lane & 15, mb = (lane >> 4) << 3;
    int nc = (qt + 2) >> 1;                  // number of 32-key chunks

    for (int c = 0; c < nc; ++c) {
        __syncthreads();
        {   // each lane loads one K row and one V row (32x64 tiles)
            size_t base = (size_t)(bb * T + c * 32 + lane) * C3 + hh * HD;
            const v4u* ksrc = (const v4u*)(qkv + base + D);
            const v4u* vsrc = (const v4u*)(qkv + base + 2 * D);
            v4u* kdst = (v4u*)(kt_ + lane * 72);
            v4u* vdst = (v4u*)(vt_ + lane * 72);
#pragma unroll
            for (int i = 0; i < 8; ++i) { kdst[i] = ksrc[i]; vdst[i] = vsrc[i]; }
        }
        __syncthreads();

        // two 16x16 score subtiles; subtile st covers keys 32c+16st..+15
        v8f s[2];
#pragma unroll
        for (int st = 0; st < 2; ++st) {
#pragma unroll
            for (int v = 0; v < 8; ++v) s[st][v] = 0.0f;
            v16bf kf = ld_frag(kt_, 72, st * 16, 0, lane);
            s[st] = __builtin_amdgcn_wmma_f32_16x16x32_bf16(
                false, qf0, false, kf, (short)0, s[st], false, false);
            kf = ld_frag(kt_, 72, st * 16, 32, lane);
            s[st] = __builtin_amdgcn_wmma_f32_16x16x32_bf16(
                false, qf1, false, kf, (short)0, s[st], false, false);

            int g = 2 * c + st;              // 16-key granule index
#pragma unroll
            for (int v = 0; v < 8; ++v) {
                float val = s[st][v] * 0.125f;            // 1/sqrt(64)
                if (g > qt || (g == qt && nn > (mb + v))) val = -3.0e38f;
                s[st][v] = val;
            }
        }

        // online softmax: shared row max over both subtiles, rescale, exp
#pragma unroll
        for (int v = 0; v < 8; ++v) {
            float x = fmaxf(s[0][v], s[1][v]);   // row lives in 16 lanes
            x = fmaxf(x, __shfl_xor(x, 1));
            x = fmaxf(x, __shfl_xor(x, 2));
            x = fmaxf(x, __shfl_xor(x, 4));
            x = fmaxf(x, __shfl_xor(x, 8));
            float mnew  = fmaxf(mst[v], x);
            float alpha = __expf(mst[v] - mnew);
            mst[v] = mnew;
            s[0][v] = __expf(s[0][v] - mnew);
            s[1][v] = __expf(s[1][v] - mnew);
#pragma unroll
            for (int j = 0; j < 5; ++j) o[j][v] *= alpha;
        }

        // stage P (16 queries x 32 keys) as bf16
#pragma unroll
        for (int v = 0; v < 8; ++v) {
            pb_[(mb + v) * 40 + nn]      = f2bf(s[0][v]);
            pb_[(mb + v) * 40 + 16 + nn] = f2bf(s[1][v]);
        }
        __syncthreads();

        // O += P V (dense K=32); row-sum accumulator += P @ ones
        v16bf pf = ld_frag(pb_, 40, 0, 0, lane);
#pragma unroll
        for (int j = 0; j < 4; ++j) {
            F16frag vf;
#pragma unroll
            for (int i = 0; i < 8; ++i) {
                int kb = (lane >> 4) << 3;
                vf.u[i]     = vt_[(kb + i) * 72      + j * 16 + nn];
                vf.u[i + 8] = vt_[(16 + kb + i) * 72 + j * 16 + nn];
            }
            o[j] = __builtin_amdgcn_wmma_f32_16x16x32_bf16(
                false, pf, false, vf.v, (short)0, o[j], false, false);
        }
        o[4] = __builtin_amdgcn_wmma_f32_16x16x32_bf16(
            false, pf, false, ones.v, (short)0, o[4], false, false);
    }

    float linv[8];
#pragma unroll
    for (int v = 0; v < 8; ++v) linv[v] = 1.0f / o[4][v];
#pragma unroll
    for (int j = 0; j < 4; ++j)
#pragma unroll
        for (int v = 0; v < 8; ++v) {
            int m = qt * 16 + mb + v;
            int d = hh * HD + j * 16 + nn;
            xatt[(size_t)(bb * T + m) * D + d] = f2bf(o[j][v] * linv[v]);
        }
}

// ---------------------------------------------------------------------------
// Host-side launch sequence
// ---------------------------------------------------------------------------
extern "C" void kernel_launch(void* const* d_in, const int* in_sizes, int n_in,
                              void* d_out, int out_size, void* d_ws,
                              size_t ws_size, hipStream_t stream) {
    (void)in_sizes; (void)n_in; (void)out_size; (void)ws_size;
    const int L = 6, Dm = 768, T = 1024, Bn = 4, V = 32000, H = 12;
    const int M = Bn * T;  // 4096

    const int*   x      = (const int*)d_in[0];
    const float* tok    = (const float*)d_in[1];
    const float* pos    = (const float*)d_in[2];
    const float* qkv_w  = (const float*)d_in[3];
    const float* fc_w   = (const float*)d_in[4];
    const float* fc_b   = (const float*)d_in[5];
    const float* ln1_s  = (const float*)d_in[6];
    const float* ln1_b  = (const float*)d_in[7];
    const float* ln2_s  = (const float*)d_in[8];
    const float* ln2_b  = (const float*)d_in[9];
    const float* ff1_w  = (const float*)d_in[10];
    const float* ff1_b  = (const float*)d_in[11];
    const float* ff2_w  = (const float*)d_in[12];
    const float* ff2_b  = (const float*)d_in[13];
    const float* lnf_s  = (const float*)d_in[14];
    const float* lnf_b  = (const float*)d_in[15];
    const float* out_w  = (const float*)d_in[16];
    const float* out_b  = (const float*)d_in[17];

    char* ws = (char*)d_ws;
    float* h = (float*)ws;                      ws += (size_t)M * Dm * 4;
    unsigned short* xln  = (unsigned short*)ws; ws += (size_t)M * Dm * 2;
    unsigned short* qkvb = (unsigned short*)ws; ws += (size_t)M * 3 * Dm * 2;
    unsigned short* xatt = (unsigned short*)ws; ws += (size_t)M * Dm * 2;
    unsigned short* ffb  = (unsigned short*)ws; // M * 4D bf16

    embed_k<<<M, 256, 0, stream>>>(x, tok, pos, h);

    for (int l = 0; l < L; ++l) {
        ln_k<<<M, 256, 0, stream>>>(h, ln1_s + l * Dm, ln1_b + l * Dm, xln);

        gemm_bf16w<2, false, false><<<dim3((3 * Dm) / BN, M / BM), 256, 0, stream>>>(
            xln, qkv_w + (size_t)l * 3 * Dm * Dm, nullptr, nullptr, qkvb, nullptr,
            3 * Dm, Dm);

        attn_k<<<Bn * H * (T / 16), 32, 0, stream>>>(qkvb, xatt);

        gemm_bf16w<1, true, false><<<dim3(Dm / BN, M / BM), 256, 0, stream>>>(
            xatt, fc_w + (size_t)l * Dm * Dm, fc_b + (size_t)l * Dm, h, nullptr, h,
            Dm, Dm);

        ln_k<<<M, 256, 0, stream>>>(h, ln2_s + l * Dm, ln2_b + l * Dm, xln);

        gemm_bf16w<2, true, true><<<dim3((4 * Dm) / BN, M / BM), 256, 0, stream>>>(
            xln, ff1_w + (size_t)l * 4 * Dm * Dm, ff1_b + (size_t)l * 4 * Dm,
            nullptr, ffb, nullptr, 4 * Dm, Dm);

        gemm_bf16w<1, true, false><<<dim3(Dm / BN, M / BM), 256, 0, stream>>>(
            ffb, ff2_w + (size_t)l * Dm * 4 * Dm, ff2_b + (size_t)l * Dm, h,
            nullptr, h, Dm, 4 * Dm);
    }

    ln_k<<<M, 256, 0, stream>>>(h, lnf_s, lnf_b, xln);

    gemm_bf16w<0, true, false><<<dim3(V / BN, M / BM), 256, 0, stream>>>(
        xln, out_w, out_b, (float*)d_out, nullptr, nullptr, V, Dm);
}
